// SwinBlock_66511863546368
// MI455X (gfx1250) — matrix operand order repaired and checked by reading, not compiled, not run on hardware
//
#include <hip/hip_runtime.h>
#include <hip/hip_bf16.h>

typedef __attribute__((ext_vector_type(16))) _Float16 v16h;
typedef __attribute__((ext_vector_type(8)))  _Float16 v8h;
typedef __attribute__((ext_vector_type(8)))  float    v8f;

#define CB     96
#define NHEADS 4
#define HDIM   24
#define NTOK   64
#define MLPH   384
#define HH     256
#define WWID   256
#define WSZ    8
#define SHIFTX 4

// f16 weight workspace offsets (in halves)
#define OFF_WQ 0          // qkv_w  [288][96]
#define OFF_WP 27648      // proj_w [96][96]
#define OFF_W1 36864      // fc1_w  [384][96]
#define OFF_W2 73728      // fc2_w  [96][384]
#define WS_HALFS 110592

static __device__ __forceinline__ v8f zero8() {
  v8f z = {0.f,0.f,0.f,0.f,0.f,0.f,0.f,0.f};
  return z;
}
static __device__ __forceinline__ v8f wmma32f16(v16h a, v16h b, v8f c) {
  // D = A(16x32 f16) * B(32x16 f16) + C(16x16 f32)
  return __builtin_amdgcn_wmma_f32_16x16x32_f16(false, a, false, b, (short)0, c,
                                                false, false);
}
// A-operand fragment: lane holds row m; halves 0..7 = K[base..base+7],
// halves 8..15 = K[base+16..base+23], base = k0 + (lane>=16 ? 8 : 0).
static __device__ __forceinline__ v16h loadA(const _Float16* p) {
  v8h lo = *(const v8h*)(p);
  v8h hi = *(const v8h*)(p + 16);
  return __builtin_shufflevector(lo, hi, 0,1,2,3,4,5,6,7,8,9,10,11,12,13,14,15);
}

// ---- CDNA5 async global<->LDS DMA (ASYNCcnt-tracked), per §15.18.3 ----
static __device__ __forceinline__ void async_g2l_b32(void* lds, const void* gaddr) {
  unsigned off = (unsigned)(size_t)lds;   // flat LDS ptr: addr[31:0] = LDS offset
  asm volatile("global_load_async_to_lds_b32 %0, %1, off"
               :: "v"(off), "v"(gaddr) : "memory");
}
static __device__ __forceinline__ void async_l2g_b32(void* gaddr, const void* lds) {
  unsigned off = (unsigned)(size_t)lds;
  asm volatile("global_store_async_from_lds_b32 %0, %1, off"
               :: "v"(gaddr), "v"(off) : "memory");
}
static __device__ __forceinline__ void wait_async0() {
  asm volatile("s_wait_asynccnt 0" ::: "memory");
}

// ---------------------------------------------------------------- weights
__global__ void wconvert(const float* __restrict__ qkv_w,
                         const float* __restrict__ proj_w,
                         const float* __restrict__ fc1_w,
                         const float* __restrict__ fc2_w,
                         _Float16* __restrict__ ws) {
  int i = blockIdx.x * 256 + threadIdx.x;
  if (i >= WS_HALFS) return;
  float v;
  if (i < OFF_WP)       v = qkv_w[i];
  else if (i < OFF_W1)  v = proj_w[i - OFF_WP];
  else if (i < OFF_W2)  v = fc1_w[i - OFF_W1];
  else                  v = fc2_w[i - OFF_W2];
  ws[i] = (_Float16)v;
}

// ------------------------------------------------- LN1 + W-MSA + residual
__global__ void __launch_bounds__(128)
swin_attn(const float* __restrict__ x,
          const float* __restrict__ norm_w, const float* __restrict__ norm_b,
          const _Float16* __restrict__ wgt,
          const float* __restrict__ qkv_b, const float* __restrict__ proj_b,
          const float* __restrict__ rpb,
          float* __restrict__ out)
{
  extern __shared__ char smem[];
  float*    xraw = (float*)smem;                     // 24576 B  raw x / residual acc
  _Float16* aln  = (_Float16*)(smem + 24576);        // 12288 B  LN'd tokens
  _Float16* qS   = (_Float16*)(smem + 36864);        // 16384 B  q [4][64][32] padded
  _Float16* kS   = (_Float16*)(smem + 53248);        // 16384 B  k [4][64][32] padded
  _Float16* vT   = (_Float16*)(smem + 69632);        // 16384 B  v^T [4][32][64]
  _Float16* pS   = (_Float16*)(smem + 86016);        // 32768 B  probs [4][64][64]
  _Float16* aout = (_Float16*)(smem + 118784);       // 12288 B  attn out [64][96]

  const int tid  = threadIdx.x;
  const int wv   = tid >> 5;
  const int lane = tid & 31;
  const int l16  = lane & 15;
  const int kh   = lane >> 4;

  const int blk = blockIdx.x;
  const int b   = blk >> 10;
  const int wi  = (blk >> 5) & 31;
  const int wj  = blk & 31;

  // ---- async DMA of shifted window (raw x) into LDS; overlap zero-fill ----
  for (int idx = tid; idx < NTOK*CB; idx += 128) {
    int c = idx >> 6, tok = idx & 63;
    int r = tok >> 3, cc = tok & 7;
    int sh = (wi*WSZ + r + SHIFTX) & (HH-1);
    int sw = (wj*WSZ + cc + SHIFTX) & (WWID-1);
    async_g2l_b32(&xraw[tok*CB + c],
                  &x[((b*CB + c) << 16) + (sh << 8) + sw]);
  }
  unsigned int* zp = (unsigned int*)qS;   // qS,kS,vT contiguous: 49152 B
  for (int idx = tid; idx < 12288; idx += 128) zp[idx] = 0u;
  wait_async0();
  __syncthreads();

  // ---- LayerNorm 1 (one thread per token) ----
  if (tid < NTOK) {
    const float* xr = xraw + tid*CB;
    float mu = 0.f;
#pragma unroll 8
    for (int c = 0; c < CB; ++c) mu += xr[c];
    mu *= (1.f/CB);
    float var = 0.f;
#pragma unroll 8
    for (int c = 0; c < CB; ++c) { float d = xr[c]-mu; var += d*d; }
    float rs = rsqrtf(var*(1.f/CB) + 1e-6f);
    _Float16* ar = aln + tid*CB;
#pragma unroll 8
    for (int c = 0; c < CB; ++c)
      ar[c] = (_Float16)((xr[c]-mu)*rs*norm_w[c] + norm_b[c]);
  }
  __syncthreads();

  // ---- QKV GEMM: [64,96] x [96,288] ----
  const _Float16* wq = wgt + OFF_WQ;
  const float qscale = 0.20412414523193154f;   // 24^-0.5
  for (int t = wv; t < 72; t += 4) {
    int mt = t & 3, nt = t >> 2;
    int o = nt*16 + l16;
    v8f acc = zero8();
#pragma unroll
    for (int kt = 0; kt < 3; ++kt) {
      int k0 = kt*32;
      v16h A  = loadA(aln + (mt*16 + l16)*CB + k0 + kh*8);
      v16h Bv = loadA(wq + o*CB + k0 + kh*8);
      acc = wmma32f16(A, Bv, acc);
    }
    float bias = qkv_b[o];
    int mbase = mt*16 + kh*8;
    if (o < 96) {
      int hd = o / HDIM, d = o % HDIM;
#pragma unroll
      for (int r = 0; r < 8; ++r)
        qS[(hd*NTOK + mbase + r)*32 + d] = (_Float16)((acc[r] + bias) * qscale);
    } else if (o < 192) {
      int oo = o - 96; int hd = oo / HDIM, d = oo % HDIM;
#pragma unroll
      for (int r = 0; r < 8; ++r)
        kS[(hd*NTOK + mbase + r)*32 + d] = (_Float16)(acc[r] + bias);
    } else {
      int oo = o - 192; int hd = oo / HDIM, d = oo % HDIM;
#pragma unroll
      for (int r = 0; r < 8; ++r)
        vT[(hd*32 + d)*NTOK + mbase + r] = (_Float16)(acc[r] + bias);
    }
  }
  __syncthreads();

  // ---- attention, one head per wave ----
  {
    const int hd = wv;
    for (int mt = 0; mt < 4; ++mt) {
      v16h A = loadA(qS + (hd*NTOK + mt*16 + l16)*32 + kh*8);
      v8f s[4];
#pragma unroll
      for (int nt = 0; nt < 4; ++nt) {
        v16h Bv = loadA(kS + (hd*NTOK + nt*16 + l16)*32 + kh*8);
        s[nt] = wmma32f16(A, Bv, zero8());
      }
      // rel-pos bias + shift mask
#pragma unroll
      for (int nt = 0; nt < 4; ++nt) {
        int n = nt*16 + l16;
        int nr = n >> 3, nc = n & 7;
        int idn = ((wi==31) ? (nr < 4 ? 1 : 2) : 0)*3
                + ((wj==31) ? (nc < 4 ? 1 : 2) : 0);
#pragma unroll
        for (int r = 0; r < 8; ++r) {
          int m = mt*16 + kh*8 + r;
          int mr = m >> 3, mc = m & 7;
          int idm = ((wi==31) ? (mr < 4 ? 1 : 2) : 0)*3
                  + ((wj==31) ? (mc < 4 ? 1 : 2) : 0);
          int rel = (mr - nr + 7)*15 + (mc - nc + 7);
          float add = rpb[rel*NHEADS + hd] + ((idm != idn) ? -100.f : 0.f);
          s[nt][r] += add;
        }
      }
      // softmax over each row (64 values spread over 4 tiles x 16 lanes)
      float rmax[8], rsum[8];
#pragma unroll
      for (int r = 0; r < 8; ++r)
        rmax[r] = fmaxf(fmaxf(s[0][r], s[1][r]), fmaxf(s[2][r], s[3][r]));
#pragma unroll
      for (int d = 1; d < 16; d <<= 1)
#pragma unroll
        for (int r = 0; r < 8; ++r)
          rmax[r] = fmaxf(rmax[r], __shfl_xor(rmax[r], d, 32));
#pragma unroll
      for (int r = 0; r < 8; ++r) rsum[r] = 0.f;
#pragma unroll
      for (int nt = 0; nt < 4; ++nt)
#pragma unroll
        for (int r = 0; r < 8; ++r) {
          float e = __expf(s[nt][r] - rmax[r]);
          s[nt][r] = e; rsum[r] += e;
        }
#pragma unroll
      for (int d = 1; d < 16; d <<= 1)
#pragma unroll
        for (int r = 0; r < 8; ++r)
          rsum[r] += __shfl_xor(rsum[r], d, 32);
#pragma unroll
      for (int r = 0; r < 8; ++r) rsum[r] = 1.f / rsum[r];
#pragma unroll
      for (int nt = 0; nt < 4; ++nt) {
        int n = nt*16 + l16;
#pragma unroll
        for (int r = 0; r < 8; ++r) {
          int m = mt*16 + kh*8 + r;
          pS[(hd*NTOK + m)*NTOK + n] = (_Float16)(s[nt][r]*rsum[r]);
        }
      }
    }
    // out_h = P[64,64] @ V[64,24]
    for (int mt = 0; mt < 4; ++mt) {
#pragma unroll
      for (int nt = 0; nt < 2; ++nt) {
        v8f acc = zero8();
#pragma unroll
        for (int kt = 0; kt < 2; ++kt) {
          int k0 = kt*32;
          v16h A  = loadA(pS + (hd*NTOK + mt*16 + l16)*NTOK + k0 + kh*8);
          v16h Bv = loadA(vT + (hd*32 + nt*16 + l16)*NTOK + k0 + kh*8);
          acc = wmma32f16(A, Bv, acc);
        }
        int d = nt*16 + l16;
        if (d < HDIM) {
          int ch = hd*HDIM + d;
#pragma unroll
          for (int r = 0; r < 8; ++r)
            aout[(mt*16 + kh*8 + r)*CB + ch] = (_Float16)acc[r];
        }
      }
    }
  }
  __syncthreads();

  // ---- proj [64,96]x[96,96] + residual into xraw ----
  const _Float16* wp = wgt + OFF_WP;
  for (int t = wv; t < 24; t += 4) {
    int mt = t & 3, nt = t >> 2;
    int o = nt*16 + l16;
    v8f acc = zero8();
#pragma unroll
    for (int kt = 0; kt < 3; ++kt) {
      int k0 = kt*32;
      v16h A  = loadA(aout + (mt*16 + l16)*CB + k0 + kh*8);
      v16h Bv = loadA(wp + o*CB + k0 + kh*8);
      acc = wmma32f16(A, Bv, acc);
    }
    float bias = proj_b[o];
#pragma unroll
    for (int r = 0; r < 8; ++r) {
      int m = mt*16 + kh*8 + r;
      xraw[m*CB + o] += acc[r] + bias;
    }
  }
  __syncthreads();

  // ---- async DMA of h back to global in final [B,C,H,W] layout ----
  for (int idx = tid; idx < NTOK*CB; idx += 128) {
    int c = idx >> 6, tok = idx & 63;
    int r = tok >> 3, cc = tok & 7;
    int sh = (wi*WSZ + r + SHIFTX) & (HH-1);
    int sw = (wj*WSZ + cc + SHIFTX) & (WWID-1);
    async_l2g_b32(&out[((b*CB + c) << 16) + (sh << 8) + sw],
                  &xraw[tok*CB + c]);
  }
  wait_async0();
}

// ------------------------------------------------- LN2 + MLP + residual
__global__ void __launch_bounds__(128)
swin_mlp(const float* __restrict__ norm2_w, const float* __restrict__ norm2_b,
         const _Float16* __restrict__ wgt,
         const float* __restrict__ fc1_b, const float* __restrict__ fc2_b,
         float* __restrict__ out)
{
  extern __shared__ char smem[];
  float*    hraw = (float*)smem;               // 24576 B
  _Float16* aln  = (_Float16*)(smem + 24576);  // 12288 B
  _Float16* hmid = (_Float16*)(smem + 36864);  // 49152 B  [64][384]

  const int tid  = threadIdx.x;
  const int wv   = tid >> 5;
  const int lane = tid & 31;
  const int l16  = lane & 15;
  const int kh   = lane >> 4;

  const int blk  = blockIdx.x;
  const int b    = blk >> 10;
  const int hrow = (blk >> 2) & 255;
  const int w0   = (blk & 3) << 6;

  // ---- async DMA of h tile into LDS ----
  for (int idx = tid; idx < NTOK*CB; idx += 128) {
    int c = idx >> 6, t = idx & 63;
    async_g2l_b32(&hraw[t*CB + c],
                  &out[((b*CB + c) << 16) + (hrow << 8) + w0 + t]);
  }
  wait_async0();
  __syncthreads();

  if (tid < NTOK) {
    const float* xr = hraw + tid*CB;
    float mu = 0.f;
#pragma unroll 8
    for (int c = 0; c < CB; ++c) mu += xr[c];
    mu *= (1.f/CB);
    float var = 0.f;
#pragma unroll 8
    for (int c = 0; c < CB; ++c) { float d = xr[c]-mu; var += d*d; }
    float rs = rsqrtf(var*(1.f/CB) + 1e-6f);
    _Float16* ar = aln + tid*CB;
#pragma unroll 8
    for (int c = 0; c < CB; ++c)
      ar[c] = (_Float16)((xr[c]-mu)*rs*norm2_w[c] + norm2_b[c]);
  }
  __syncthreads();

  // fc1: [64,96] x [96,384] + exact GELU
  const _Float16* w1 = wgt + OFF_W1;
  for (int t = wv; t < 96; t += 4) {
    int mt = t & 3, nt = t >> 2;
    int o = nt*16 + l16;
    v8f acc = zero8();
#pragma unroll
    for (int kt = 0; kt < 3; ++kt) {
      int k0 = kt*32;
      v16h A  = loadA(aln + (mt*16 + l16)*CB + k0 + kh*8);
      v16h Bv = loadA(w1 + o*CB + k0 + kh*8);
      acc = wmma32f16(A, Bv, acc);
    }
    float bias = fc1_b[o];
#pragma unroll
    for (int r = 0; r < 8; ++r) {
      float v = acc[r] + bias;
      float g = 0.5f * v * (1.f + erff(v * 0.70710678118654752f));
      hmid[(mt*16 + kh*8 + r)*MLPH + o] = (_Float16)g;
    }
  }
  __syncthreads();

  // fc2: [64,384] x [384,96] + residual
  const _Float16* w2 = wgt + OFF_W2;
  for (int t = wv; t < 24; t += 4) {
    int mt = t & 3, nt = t >> 2;
    int o = nt*16 + l16;
    v8f acc = zero8();
#pragma unroll
    for (int kt = 0; kt < 12; ++kt) {
      int k0 = kt*32;
      v16h A  = loadA(hmid + (mt*16 + l16)*MLPH + k0 + kh*8);
      v16h Bv = loadA(w2 + o*MLPH + k0 + kh*8);
      acc = wmma32f16(A, Bv, acc);
    }
    float bias = fc2_b[o];
#pragma unroll
    for (int r = 0; r < 8; ++r) {
      int m = mt*16 + kh*8 + r;
      hraw[m*CB + o] += acc[r] + bias;
    }
  }
  __syncthreads();

  // ---- async DMA of final tile back to global ----
  for (int idx = tid; idx < NTOK*CB; idx += 128) {
    int c = idx >> 6, t = idx & 63;
    async_l2g_b32(&out[((b*CB + c) << 16) + (hrow << 8) + w0 + t],
                  &hraw[t*CB + c]);
  }
  wait_async0();
}

// ----------------------------------------------------------------------
extern "C" void kernel_launch(void* const* d_in, const int* in_sizes, int n_in,
                              void* d_out, int out_size, void* d_ws, size_t ws_size,
                              hipStream_t stream) {
  const float* x      = (const float*)d_in[0];
  const float* norm_w = (const float*)d_in[1];
  const float* norm_b = (const float*)d_in[2];
  const float* qkv_w  = (const float*)d_in[3];
  const float* qkv_b  = (const float*)d_in[4];
  const float* proj_w = (const float*)d_in[5];
  const float* proj_b = (const float*)d_in[6];
  const float* rpb    = (const float*)d_in[7];
  const float* n2w    = (const float*)d_in[8];
  const float* n2b    = (const float*)d_in[9];
  const float* fc1_w  = (const float*)d_in[10];
  const float* fc1_b  = (const float*)d_in[11];
  const float* fc2_w  = (const float*)d_in[12];
  const float* fc2_b  = (const float*)d_in[13];
  float* out = (float*)d_out;
  _Float16* wgt = (_Float16*)d_ws;

  (void)in_sizes; (void)n_in; (void)out_size; (void)ws_size;

  hipFuncSetAttribute((const void*)swin_attn,
                      hipFuncAttributeMaxDynamicSharedMemorySize, 131072);
  hipFuncSetAttribute((const void*)swin_mlp,
                      hipFuncAttributeMaxDynamicSharedMemorySize, 86016);

  wconvert<<<(WS_HALFS + 255) / 256, 256, 0, stream>>>(qkv_w, proj_w, fc1_w, fc2_w, wgt);
  swin_attn<<<4096, 128, 131072, stream>>>(x, norm_w, norm_b, wgt, qkv_b, proj_b, rpb, out);
  swin_mlp<<<4096, 128, 86016, stream>>>(n2w, n2b, wgt, fc1_b, fc2_b, out);
}
